// BipartiteGATExtended_33603824124608
// MI455X (gfx1250) — compile-verified, matrix operand order checked by reading, not run on hardware
//
#include <hip/hip_runtime.h>

typedef __attribute__((ext_vector_type(16))) _Float16 v16h;
typedef __attribute__((ext_vector_type(8)))  float    v8f;

constexpr int NPOL  = 20000;
constexpr int NCOMP = 80000;
constexpr int NN    = 100000;          // NPOL + NCOMP
constexpr int EG    = 1000000;         // real edges
constexpr int ET    = 1100000;         // + self loops
constexpr int EMB   = 128;
constexpr int HID   = 256;
constexpr int OUTD  = 128;
constexpr int HEADS = 8;

// ---------------- workspace arena (bytes) ----------------
constexpr size_t A256(size_t x) { return (x + 255) & ~(size_t)255; }
constexpr size_t OFF_XH     = 0;                                        // f16 [NN,128]  25.6MB (dead after GEMM1)
constexpr size_t OFF_XW1    = OFF_XH     + A256((size_t)NN * EMB * 2);  // f32 [NN,256] 102.4MB
constexpr size_t OFF_OUT1   = OFF_XW1    + A256((size_t)NN * HID * 4);  // f32 [NN,256] 102.4MB
constexpr size_t OFF_LOGIT1 = OFF_OUT1   + A256((size_t)NN * HID * 4);  // f32 [ET,8]    35.2MB
constexpr size_t OFF_ALS1   = OFF_LOGIT1 + A256((size_t)ET * 8 * 4);
constexpr size_t OFF_ALD1   = OFF_ALS1   + A256((size_t)NN * 8 * 4);
constexpr size_t OFF_AMAX1  = OFF_ALD1   + A256((size_t)NN * 8 * 4);
constexpr size_t OFF_DEN1   = OFF_AMAX1  + A256((size_t)NN * 8 * 4);
constexpr size_t OFF_LOOPA  = OFF_DEN1   + A256((size_t)NN * 8 * 4);    // f32 [NN,5]
constexpr size_t OFF_CNT    = OFF_LOOPA  + A256((size_t)NN * 5 * 4);    // f32 [NN]
constexpr size_t OFF_W1T    = OFF_CNT    + A256((size_t)NN * 4);        // f16 [256,128]
constexpr size_t OFF_W2T    = OFF_W1T    + A256((size_t)256 * 128 * 2); // f16 [128,256]
constexpr size_t OFF_M1     = OFF_W2T    + A256((size_t)128 * 256 * 2); // f32 [5,8]
constexpr size_t OFF_M2     = OFF_M1     + 256;                         // f32 [5]
// phase-2 reuse (lifetimes disjoint):
constexpr size_t OFF_HH     = OFF_XW1;   // f16 [NN,256] over dead xw1
constexpr size_t OFF_XW2    = OFF_OUT1;  // f32 [NN,128] over dead out1
constexpr size_t OFF_ALS2   = OFF_XH;                // f32 [NN]  (xh dead)
constexpr size_t OFF_ALD2   = OFF_XH +  524288;
constexpr size_t OFF_AMAX2  = OFF_XH + 1048576;
constexpr size_t OFF_DEN2   = OFF_XH + 1572864;
constexpr size_t OFF_LOGIT2 = OFF_XH + 2097152;      // f32 [ET]

// ---------------- small helpers ----------------
__device__ __forceinline__ float wave_sum32(float v) {
#pragma unroll
  for (int m = 16; m >= 1; m >>= 1) v += __shfl_xor(v, m, 32);
  return v;
}

__device__ __forceinline__ void atomicMaxF(float* addr, float val) {
  unsigned int* ua  = (unsigned int*)addr;
  unsigned int  cur = *(volatile unsigned int*)addr;
  while (__uint_as_float(cur) < val) {
    unsigned int prev = atomicCAS(ua, cur, __float_as_uint(val));
    if (prev == cur) break;
    cur = prev;
  }
}

__global__ void k_fill(float* p, float v, int n) {
  int i = blockIdx.x * blockDim.x + threadIdx.x;
  if (i < n) p[i] = v;
}

// ---------------- prep: weight transpose to f16, fold edge-attention matrices ----------------
__global__ void k_w1t(const float* __restrict__ W1, _Float16* __restrict__ W1t) {
  int i = blockIdx.x * blockDim.x + threadIdx.x;              // 256*128
  if (i >= 256 * 128) return;
  int n = i >> 7, k = i & 127;                                // W1 [128,256] -> W1t [256,128]
  W1t[i] = (_Float16)W1[k * 256 + n];
}
__global__ void k_w2t(const float* __restrict__ W2, _Float16* __restrict__ W2t) {
  int i = blockIdx.x * blockDim.x + threadIdx.x;              // 128*256
  if (i >= 128 * 256) return;
  int n = i >> 8, k = i & 255;                                // W2 [256,128] -> W2t [128,256]
  W2t[i] = (_Float16)W2[k * 128 + n];
}
__global__ void k_fold_edge_att(const float* __restrict__ We1, const float* __restrict__ ae1,
                                const float* __restrict__ We2, const float* __restrict__ ae2,
                                float* __restrict__ M1, float* __restrict__ M2) {
  int t = blockIdx.x * blockDim.x + threadIdx.x;
  if (t < 40) {                       // M1[k,h] = sum_c We1[k,h*32+c]*ae1[h,c]
    int k = t >> 3, h = t & 7;
    float s = 0.f;
    for (int c = 0; c < 32; ++c) s += We1[k * 256 + h * 32 + c] * ae1[h * 32 + c];
    M1[k * 8 + h] = s;
  } else if (t < 45) {                // M2[k] = sum_c We2[k,c]*ae2[c]
    int k = t - 40;
    float s = 0.f;
    for (int c = 0; c < 128; ++c) s += We2[k * 128 + c] * ae2[c];
    M2[k] = s;
  }
}

// ---------------- self-loop mean edge-attrs ----------------
__global__ void k_loop_accum(const int* __restrict__ ei, const float* __restrict__ ea,
                             float* __restrict__ loopa, float* __restrict__ cnt) {
  int e = blockIdx.x * blockDim.x + threadIdx.x;
  if (e >= EG) return;
  int d = ei[EG + e];
  atomicAdd(&cnt[d], 1.0f);
#pragma unroll
  for (int k = 0; k < 5; ++k) atomicAdd(&loopa[d * 5 + k], ea[e * 5 + k]);
}
__global__ void k_loop_final(float* __restrict__ loopa, const float* __restrict__ cnt) {
  int n = blockIdx.x * blockDim.x + threadIdx.x;
  if (n >= NN) return;
  float inv = 1.0f / fmaxf(cnt[n], 1.0f);
#pragma unroll
  for (int k = 0; k < 5; ++k) loopa[n * 5 + k] *= inv;
}

// ---------------- fused encoders + LayerNorm -> f16 ----------------
__global__ void k_encode(const float* __restrict__ pol_feat, const int* __restrict__ state_ids,
                         const int* __restrict__ comp_cat, const float* __restrict__ comp_price,
                         const float* __restrict__ Wp, const float* __restrict__ bp,
                         const float* __restrict__ state_emb, const float* __restrict__ sector_emb,
                         const float* __restrict__ industry_emb, const float* __restrict__ Wc,
                         const float* __restrict__ bc, const float* __restrict__ ln_g,
                         const float* __restrict__ ln_b, _Float16* __restrict__ xh) {
  int wid  = (blockIdx.x * blockDim.x + threadIdx.x) >> 5;    // one wave per node
  int lane = threadIdx.x & 31;
  if (wid >= NN) return;
  int c0 = lane * 4;
  float acc[4];
  if (wid < NPOL) {
#pragma unroll
    for (int j = 0; j < 4; ++j) acc[j] = bp[c0 + j];
    const float* pf = pol_feat + wid * 7;
    for (int k = 0; k < 7; ++k) {
      float f = pf[k];
      const float* wr = Wp + k * EMB + c0;
#pragma unroll
      for (int j = 0; j < 4; ++j) acc[j] += f * wr[j];
    }
    const float* se = state_emb + state_ids[wid] * EMB + c0;
#pragma unroll
    for (int j = 0; j < 4; ++j) acc[j] = fmaxf(acc[j], 0.f) + se[j];
  } else {
    int m = wid - NPOL;
#pragma unroll
    for (int j = 0; j < 4; ++j) acc[j] = bc[c0 + j];
    int sid = comp_cat[m * 2 + 0], iid = comp_cat[m * 2 + 1];
    for (int k = 0; k < 8; ++k) {
      float f = sector_emb[sid * 8 + k];
      const float* wr = Wc + k * EMB + c0;
#pragma unroll
      for (int j = 0; j < 4; ++j) acc[j] += f * wr[j];
    }
    for (int k = 0; k < 8; ++k) {
      float f = industry_emb[iid * 8 + k];
      const float* wr = Wc + (8 + k) * EMB + c0;
#pragma unroll
      for (int j = 0; j < 4; ++j) acc[j] += f * wr[j];
    }
    float pr = comp_price[m];
    const float* wr = Wc + 16 * EMB + c0;
#pragma unroll
    for (int j = 0; j < 4; ++j) acc[j] = fmaxf(acc[j] + pr * wr[j], 0.f);
  }
  float s = 0.f, ss = 0.f;
#pragma unroll
  for (int j = 0; j < 4; ++j) { s += acc[j]; ss += acc[j] * acc[j]; }
  s  = wave_sum32(s);
  ss = wave_sum32(ss);
  float mean = s * (1.0f / EMB);
  float var  = ss * (1.0f / EMB) - mean * mean;
  float rstd = rsqrtf(var + 1e-5f);
#pragma unroll
  for (int j = 0; j < 4; ++j)
    xh[(size_t)wid * EMB + c0 + j] =
        (_Float16)((acc[j] - mean) * rstd * ln_g[c0 + j] + ln_b[c0 + j]);
}

// ---------------- WMMA GEMM: C[M,NCOLS] = A[M,K](f16) x Bt[NCOLS,K](f16), f32 accum ----------------
// CDNA5 16-bit A/B fragment: lane&15 = row/col index, lane<16 -> K-halves {0..7,16..23}, else {8..15,24..31}
__device__ __forceinline__ v16h load_frag16(const _Float16* __restrict__ p) {
  v16h a;
  ((uint4*)&a)[0] = *(const uint4*)p;
  ((uint4*)&a)[1] = *(const uint4*)(p + 16);
  return a;
}

template <int K, int NCOLS>
__global__ void k_gemm_wmma(const _Float16* __restrict__ A, const _Float16* __restrict__ Bt,
                            float* __restrict__ C, int M) {
  int wid  = (blockIdx.x * blockDim.x + threadIdx.x) >> 5;
  int lane = threadIdx.x & 31;
  constexpr int NG = NCOLS / 64;                 // each wave computes a 16x64 tile
  int mt = wid / NG, ng = wid % NG;
  int m0 = mt * 16, n0 = ng * 64;
  if (m0 >= M) return;
  int koff = (lane < 16) ? 0 : 8;
  int arow = m0 + (lane & 15);
  v8f acc[4] = {v8f{}, v8f{}, v8f{}, v8f{}};
  for (int k0 = 0; k0 < K; k0 += 32) {
    v16h a = load_frag16(A + (size_t)arow * K + k0 + koff);
#pragma unroll
    for (int j = 0; j < 4; ++j) {
      v16h b = load_frag16(Bt + (size_t)(n0 + j * 16 + (lane & 15)) * K + k0 + koff);
      acc[j] = __builtin_amdgcn_wmma_f32_16x16x32_f16(false, a, false, b,
                                                      (short)0, acc[j], false, false);
    }
  }
  int rbase = m0 + ((lane < 16) ? 0 : 8);        // D layout: VGPR r -> row r (+8 for hi lanes)
#pragma unroll
  for (int j = 0; j < 4; ++j) {
    int col = n0 + j * 16 + (lane & 15);
#pragma unroll
    for (int r = 0; r < 8; ++r)
      C[(size_t)(rbase + r) * NCOLS + col] = acc[j][r];
  }
}

// ---------------- per-node attention logits ----------------
__global__ void k_node_logits1(const float* __restrict__ xw, const float* __restrict__ asrc,
                               const float* __restrict__ adst, float* __restrict__ als,
                               float* __restrict__ ald) {
  int wid  = (blockIdx.x * blockDim.x + threadIdx.x) >> 5;
  int lane = threadIdx.x & 31;
  if (wid >= NN) return;
  const float* row = xw + (size_t)wid * HID + lane * 8;       // 8 ch/lane, head = lane/4
  float ps = 0.f, pd = 0.f;
#pragma unroll
  for (int j = 0; j < 8; ++j) {
    float v = row[j];
    ps += v * asrc[lane * 8 + j];
    pd += v * adst[lane * 8 + j];
  }
  ps += __shfl_xor(ps, 1, 32); ps += __shfl_xor(ps, 2, 32);
  pd += __shfl_xor(pd, 1, 32); pd += __shfl_xor(pd, 2, 32);
  if ((lane & 3) == 0) {
    als[wid * HEADS + (lane >> 2)] = ps;
    ald[wid * HEADS + (lane >> 2)] = pd;
  }
}

__global__ void k_node_logits2(const float* __restrict__ xw, const float* __restrict__ asrc,
                               const float* __restrict__ adst, float* __restrict__ als,
                               float* __restrict__ ald) {
  int wid  = (blockIdx.x * blockDim.x + threadIdx.x) >> 5;
  int lane = threadIdx.x & 31;
  if (wid >= NN) return;
  const float* row = xw + (size_t)wid * OUTD + lane * 4;
  float ps = 0.f, pd = 0.f;
#pragma unroll
  for (int j = 0; j < 4; ++j) {
    float v = row[j];
    ps += v * asrc[lane * 4 + j];
    pd += v * adst[lane * 4 + j];
  }
  ps = wave_sum32(ps);
  pd = wave_sum32(pd);
  if (lane == 0) { als[wid] = ps; ald[wid] = pd; }
}

// ---------------- edge passes ----------------
__device__ __forceinline__ void resolve_edge(int e, const int* __restrict__ ei,
                                             const float* __restrict__ ea,
                                             const float* __restrict__ loopa,
                                             int& s, int& d, float ev[5]) {
  if (e < EG) {
    s = ei[e]; d = ei[EG + e];
#pragma unroll
    for (int k = 0; k < 5; ++k) ev[k] = ea[(size_t)e * 5 + k];
  } else {
    s = d = e - EG;
#pragma unroll
    for (int k = 0; k < 5; ++k) ev[k] = loopa[(size_t)(e - EG) * 5 + k];
  }
}

__global__ void k_edge_max1(const int* __restrict__ ei, const float* __restrict__ ea,
                            const float* __restrict__ loopa, const float* __restrict__ als,
                            const float* __restrict__ ald, const float* __restrict__ M1,
                            float* __restrict__ logit, float* __restrict__ amax) {
  int e = blockIdx.x * blockDim.x + threadIdx.x;
  if (e >= ET) return;
  int s, d; float ev[5];
  resolve_edge(e, ei, ea, loopa, s, d, ev);
#pragma unroll
  for (int h = 0; h < HEADS; ++h) {
    float al = 0.f;
#pragma unroll
    for (int k = 0; k < 5; ++k) al += ev[k] * M1[k * 8 + h];
    float z = als[s * HEADS + h] + ald[d * HEADS + h] + al;
    z = (z > 0.f) ? z : 0.2f * z;                  // leaky_relu
    logit[(size_t)e * HEADS + h] = z;
    atomicMaxF(&amax[d * HEADS + h], z);
  }
}

__global__ void k_edge_exp1(const int* __restrict__ ei, float* __restrict__ logit,
                            const float* __restrict__ amax, float* __restrict__ denom) {
  int e = blockIdx.x * blockDim.x + threadIdx.x;
  if (e >= ET) return;
  int d = (e < EG) ? ei[EG + e] : (e - EG);
#pragma unroll
  for (int h = 0; h < HEADS; ++h) {
    float ex = __expf(logit[(size_t)e * HEADS + h] - amax[d * HEADS + h]);
    logit[(size_t)e * HEADS + h] = ex;             // reuse buffer as ex
    atomicAdd(&denom[d * HEADS + h], ex);
  }
}

__global__ void k_agg1(const int* __restrict__ ei, const float* __restrict__ xw,
                       const float* __restrict__ ex, const float* __restrict__ denom,
                       float* __restrict__ out) {
  int wid  = (blockIdx.x * blockDim.x + threadIdx.x) >> 5;    // one wave per edge
  int lane = threadIdx.x & 31;
  if (wid >= ET) return;
  int s = (wid < EG) ? ei[wid] : (wid - EG);
  int d = (wid < EG) ? ei[EG + wid] : (wid - EG);
  const float* srow = xw + (size_t)s * HID + lane * 8;
  __builtin_prefetch(srow, 0, 0);                  // global_prefetch_b8
  int h = lane >> 2;                               // 8 ch/lane -> head = lane/4
  float w = ex[(size_t)wid * HEADS + h] / (denom[d * HEADS + h] + 1e-16f);
  float4 x0 = *(const float4*)(srow);
  float4 x1 = *(const float4*)(srow + 4);
  float* drow = out + (size_t)d * HID + lane * 8;
  atomicAdd(&drow[0], x0.x * w); atomicAdd(&drow[1], x0.y * w);
  atomicAdd(&drow[2], x0.z * w); atomicAdd(&drow[3], x0.w * w);
  atomicAdd(&drow[4], x1.x * w); atomicAdd(&drow[5], x1.y * w);
  atomicAdd(&drow[6], x1.z * w); atomicAdd(&drow[7], x1.w * w);
}

__global__ void k_elu_hh(const float* __restrict__ out1, const float* __restrict__ b1,
                         _Float16* __restrict__ hh) {
  int i = blockIdx.x * blockDim.x + threadIdx.x;
  if (i >= NN * HID) return;
  float v = out1[i] + b1[i & 255];
  v = (v > 0.f) ? v : expm1f(v);                   // elu, alpha=1
  hh[i] = (_Float16)v;
}

__global__ void k_edge_max2(const int* __restrict__ ei, const float* __restrict__ ea,
                            const float* __restrict__ loopa, const float* __restrict__ als,
                            const float* __restrict__ ald, const float* __restrict__ M2,
                            float* __restrict__ logit, float* __restrict__ amax) {
  int e = blockIdx.x * blockDim.x + threadIdx.x;
  if (e >= ET) return;
  int s, d; float ev[5];
  resolve_edge(e, ei, ea, loopa, s, d, ev);
  float al = 0.f;
#pragma unroll
  for (int k = 0; k < 5; ++k) al += ev[k] * M2[k];
  float z = als[s] + ald[d] + al;
  z = (z > 0.f) ? z : 0.2f * z;
  logit[e] = z;
  atomicMaxF(&amax[d], z);
}

__global__ void k_edge_exp2(const int* __restrict__ ei, float* __restrict__ logit,
                            const float* __restrict__ amax, float* __restrict__ denom) {
  int e = blockIdx.x * blockDim.x + threadIdx.x;
  if (e >= ET) return;
  int d = (e < EG) ? ei[EG + e] : (e - EG);
  float ex = __expf(logit[e] - amax[d]);
  logit[e] = ex;
  atomicAdd(&denom[d], ex);
}

__global__ void k_init_out(float* __restrict__ out, const float* __restrict__ b2) {
  int i = blockIdx.x * blockDim.x + threadIdx.x;
  if (i < NN * OUTD) out[i] = b2[i & 127];
}

__global__ void k_agg2(const int* __restrict__ ei, const float* __restrict__ xw,
                       const float* __restrict__ ex, const float* __restrict__ denom,
                       float* __restrict__ out) {
  int wid  = (blockIdx.x * blockDim.x + threadIdx.x) >> 5;
  int lane = threadIdx.x & 31;
  if (wid >= ET) return;
  int s = (wid < EG) ? ei[wid] : (wid - EG);
  int d = (wid < EG) ? ei[EG + wid] : (wid - EG);
  const float* srow = xw + (size_t)s * OUTD + lane * 4;
  __builtin_prefetch(srow, 0, 0);
  float w = ex[wid] / (denom[d] + 1e-16f);
  float4 x = *(const float4*)(srow);
  float* drow = out + (size_t)d * OUTD + lane * 4;
  atomicAdd(&drow[0], x.x * w); atomicAdd(&drow[1], x.y * w);
  atomicAdd(&drow[2], x.z * w); atomicAdd(&drow[3], x.w * w);
}

// ---------------- launcher ----------------
extern "C" void kernel_launch(void* const* d_in, const int* in_sizes, int n_in,
                              void* d_out, int out_size, void* d_ws, size_t ws_size,
                              hipStream_t stream) {
  const float* pol_feat     = (const float*)d_in[0];
  const int*   state_ids    = (const int*)  d_in[1];
  const int*   comp_cat     = (const int*)  d_in[2];
  const float* comp_price   = (const float*)d_in[3];
  const int*   edge_index   = (const int*)  d_in[4];
  const float* edge_attr    = (const float*)d_in[5];
  const float* Wp           = (const float*)d_in[6];
  const float* bp           = (const float*)d_in[7];
  const float* state_emb    = (const float*)d_in[8];
  const float* sector_emb   = (const float*)d_in[9];
  const float* industry_emb = (const float*)d_in[10];
  const float* Wc           = (const float*)d_in[11];
  const float* bc           = (const float*)d_in[12];
  const float* ln_g         = (const float*)d_in[13];
  const float* ln_b         = (const float*)d_in[14];
  const float* W1           = (const float*)d_in[15];
  const float* att_src1     = (const float*)d_in[16];
  const float* att_dst1     = (const float*)d_in[17];
  const float* We1          = (const float*)d_in[18];
  const float* att_e1       = (const float*)d_in[19];
  const float* b1           = (const float*)d_in[20];
  const float* W2           = (const float*)d_in[21];
  const float* att_src2     = (const float*)d_in[22];
  const float* att_dst2     = (const float*)d_in[23];
  const float* We2          = (const float*)d_in[24];
  const float* att_e2       = (const float*)d_in[25];
  const float* b2           = (const float*)d_in[26];

  char* ws = (char*)d_ws;
  _Float16* xh    = (_Float16*)(ws + OFF_XH);
  float*    xw1   = (float*)(ws + OFF_XW1);
  float*    out1  = (float*)(ws + OFF_OUT1);
  float*    lg1   = (float*)(ws + OFF_LOGIT1);
  float*    als1  = (float*)(ws + OFF_ALS1);
  float*    ald1  = (float*)(ws + OFF_ALD1);
  float*    amax1 = (float*)(ws + OFF_AMAX1);
  float*    den1  = (float*)(ws + OFF_DEN1);
  float*    loopa = (float*)(ws + OFF_LOOPA);
  float*    cnt   = (float*)(ws + OFF_CNT);
  _Float16* W1t   = (_Float16*)(ws + OFF_W1T);
  _Float16* W2t   = (_Float16*)(ws + OFF_W2T);
  float*    M1    = (float*)(ws + OFF_M1);
  float*    M2    = (float*)(ws + OFF_M2);
  _Float16* hh    = (_Float16*)(ws + OFF_HH);
  float*    xw2   = (float*)(ws + OFF_XW2);
  float*    als2  = (float*)(ws + OFF_ALS2);
  float*    ald2  = (float*)(ws + OFF_ALD2);
  float*    amax2 = (float*)(ws + OFF_AMAX2);
  float*    den2  = (float*)(ws + OFF_DEN2);
  float*    lg2   = (float*)(ws + OFF_LOGIT2);
  float*    outp  = (float*)d_out;

  const int B = 256;
  auto cdiv = [](long long a, long long b) { return (unsigned)((a + b - 1) / b); };

  // prep
  k_fill<<<cdiv(NN * 5, B), B, 0, stream>>>(loopa, 0.f, NN * 5);
  k_fill<<<cdiv(NN, B), B, 0, stream>>>(cnt, 0.f, NN);
  k_w1t<<<cdiv(256 * 128, B), B, 0, stream>>>(W1, W1t);
  k_w2t<<<cdiv(128 * 256, B), B, 0, stream>>>(W2, W2t);
  k_fold_edge_att<<<1, 64, 0, stream>>>(We1, att_e1, We2, att_e2, M1, M2);
  k_loop_accum<<<cdiv(EG, B), B, 0, stream>>>(edge_index, edge_attr, loopa, cnt);
  k_loop_final<<<cdiv(NN, B), B, 0, stream>>>(loopa, cnt);

  // encode + LN
  k_encode<<<cdiv((long long)NN * 32, B), B, 0, stream>>>(
      pol_feat, state_ids, comp_cat, comp_price, Wp, bp, state_emb,
      sector_emb, industry_emb, Wc, bc, ln_g, ln_b, xh);

  // layer 1
  {
    long long waves = (long long)(NN / 16) * (HID / 64);     // 25000
    k_gemm_wmma<EMB, HID><<<cdiv(waves * 32, B), B, 0, stream>>>(xh, W1t, xw1, NN);
  }
  k_node_logits1<<<cdiv((long long)NN * 32, B), B, 0, stream>>>(xw1, att_src1, att_dst1, als1, ald1);
  k_fill<<<cdiv(NN * HEADS, B), B, 0, stream>>>(amax1, -1e30f, NN * HEADS);
  k_fill<<<cdiv(NN * HEADS, B), B, 0, stream>>>(den1, 0.f, NN * HEADS);
  k_fill<<<cdiv((long long)NN * HID, B), B, 0, stream>>>(out1, 0.f, NN * HID);
  k_edge_max1<<<cdiv(ET, B), B, 0, stream>>>(edge_index, edge_attr, loopa, als1, ald1, M1, lg1, amax1);
  k_edge_exp1<<<cdiv(ET, B), B, 0, stream>>>(edge_index, lg1, amax1, den1);
  k_agg1<<<cdiv((long long)ET * 32, B), B, 0, stream>>>(edge_index, xw1, lg1, den1, out1);
  k_elu_hh<<<cdiv((long long)NN * HID, B), B, 0, stream>>>(out1, b1, hh);

  // layer 2
  {
    long long waves = (long long)(NN / 16) * (OUTD / 64);    // 12500
    k_gemm_wmma<HID, OUTD><<<cdiv(waves * 32, B), B, 0, stream>>>(hh, W2t, xw2, NN);
  }
  k_node_logits2<<<cdiv((long long)NN * 32, B), B, 0, stream>>>(xw2, att_src2, att_dst2, als2, ald2);
  k_fill<<<cdiv(NN, B), B, 0, stream>>>(amax2, -1e30f, NN);
  k_fill<<<cdiv(NN, B), B, 0, stream>>>(den2, 0.f, NN);
  k_edge_max2<<<cdiv(ET, B), B, 0, stream>>>(edge_index, edge_attr, loopa, als2, ald2, M2, lg2, amax2);
  k_edge_exp2<<<cdiv(ET, B), B, 0, stream>>>(edge_index, lg2, amax2, den2);
  k_init_out<<<cdiv((long long)NN * OUTD, B), B, 0, stream>>>(outp, b2);
  k_agg2<<<cdiv((long long)ET * 32, B), B, 0, stream>>>(edge_index, xw2, lg2, den2, outp);
}
